// HopAggLayer2_39676907888227
// MI455X (gfx1250) — compile-verified, matrix operand order checked by reading, not run on hardware
//
#include <hip/hip_runtime.h>
#include <stdint.h>

// ---------------------------------------------------------------------------
// Problem constants (from the reference): B=16384, D=256, T=100, E=172, K=2, N=20
// ---------------------------------------------------------------------------
#define B_   16384
#define D_   256
#define T_   100
#define E_   172
#define KHOP 2
#define NNB  20
#define ROWS_NBR (B_ * KHOP * NNB)   // 655360 neighbor rows
#define KC_NBR  544                  // 528 (=256+100+172) padded to 17*32
#define NCH_NBR 17
#define KC_SRC  384                  // 356 (=256+100) padded to 12*32
#define NCH_SRC 12

typedef __attribute__((ext_vector_type(16))) __bf16 v16bf;
typedef __attribute__((ext_vector_type(8)))  float  v8f;
typedef __attribute__((ext_vector_type(4)))  unsigned v4u;
typedef __attribute__((ext_vector_type(8)))  int      v8i;
typedef __attribute__((ext_vector_type(4)))  int      v4i;

union FragU { uint4 q[2]; v16bf v; };

static __device__ __forceinline__ unsigned short f2bf(float f) {
  union { float f; unsigned u; } x; x.f = f;
  unsigned u = x.u;
  u += 0x7fffu + ((u >> 16) & 1u);   // round-to-nearest-even
  return (unsigned short)(u >> 16);
}

// Wave-relative LDS byte address of a __shared__ object: the flat shared
// aperture keeps the block-relative LDS offset in addr[31:0] (ISA 10.2).
static __device__ __forceinline__ unsigned lds_addr32(const void* p) {
  return (unsigned)(size_t)p;
}

// ---------------------------------------------------------------------------
// Kernel 0: transpose + pad + bf16-convert both weight matrices into ws.
// Layout: WT[c][kp]  (each output column's K-vector contiguous) so the GEMM
// kernels' B-slab fill is a contiguous 64B-per-column run per K-chunk.
// ---------------------------------------------------------------------------
__global__ __launch_bounds__(256)
void prep_weights(const float* __restrict__ Wn, const float* __restrict__ Ws,
                  unsigned short* __restrict__ WnT, unsigned short* __restrict__ WsT) {
  int idx = blockIdx.x * blockDim.x + threadIdx.x;
  const int n1 = D_ * KC_NBR;
  const int nTot = n1 + D_ * KC_SRC;
  if (idx >= nTot) return;
  if (idx < n1) {
    int c = idx / KC_NBR, kp = idx % KC_NBR;
    float v = (kp < (D_ + T_ + E_)) ? Wn[kp * D_ + c] : 0.f;
    WnT[idx] = f2bf(v);
  } else {
    int i = idx - n1;
    int c = i / KC_SRC, kp = i % KC_SRC;
    float v = (kp < (D_ + T_)) ? Ws[kp * D_ + c] : 0.f;
    WsT[i] = f2bf(v);
  }
}

// ---------------------------------------------------------------------------
// Kernel 1: neighbor projection GEMM + residual + mask + fused sum over N=20.
// Block = 320 threads = 10 wave32; wave w owns rows [16w,16w+16) of a
// 160-row tile (= exactly 8 (b,hop) groups).  16 column tiles of 16 cover D=256.
// B slab is brought in by the Tensor Data Mover: one 2D-tile descriptor
// (256 cols x 64B, row stride 1088B) issued by wave 0, tracked on TENSORcnt,
// overlapped with the VALU gather/convert of the A slab.
// ---------------------------------------------------------------------------
__global__ __launch_bounds__(320)
void nbr_gemm(const float* __restrict__ nf, const float* __restrict__ tf,
              const float* __restrict__ ef, const unsigned char* __restrict__ mask,
              const unsigned short* __restrict__ WnT,
              const float* __restrict__ bnbr,
              float* __restrict__ outSum /* [B*KHOP, 256] */) {
  __shared__ __align__(16) unsigned short Asl[160 * 32];  // 10.0 KB
  __shared__ __align__(16) unsigned short Bsl[256 * 32];  // 16.0 KB
  __shared__ float Esl[160 * 16];                         // 10.0 KB

  const int t    = threadIdx.x;
  const int lane = t & 31;
  const int wave = t >> 5;
  const long rowBlk = (long)blockIdx.x * 160;

  v8f acc[16] = {};   // 16 col-tiles x 16x16 f32 accumulators (128 VGPRs)

  // A-fill mapping: 2 threads per row, each covers 16 of the 32 k's.
  const int r    = t >> 1;          // 0..159
  const int half = (t & 1) * 16;    // 0 or 16
  const long R   = rowBlk + r;
  const float* nfR = nf + R * D_;
  const float* tfR = tf + R * T_;
  const float* efR = ef + R * E_;

  const unsigned bslBase = lds_addr32(&Bsl[0]);
  const unsigned long long wnBase = (unsigned long long)WnT;

  // TDM D# group1 (constant across chunks): data_size=2B, tensor_dim0=544,
  // tensor_dim1=256, tile_dim0=32, tile_dim1=256, tensor_dim0_stride=544.
  const v8i g1 = { 0x00010000, 0x02200000, 0x01000000, 0x00200000,
                   256,        544,        0,          0 };
  const v4i gz4 = { 0, 0, 0, 0 };
  const v8i gz8 = { 0, 0, 0, 0, 0, 0, 0, 0 };

  for (int ch = 0; ch < NCH_NBR; ++ch) {
    __syncthreads();
    // ---- issue the B-slab DMA (one descriptor; EXEC-independent, wave 0 only)
    if (t < 32) {
      const unsigned long long g = wnBase + (unsigned long long)(ch * 64);
      const v4u g0 = { 1u,                                   // count=1
                       bslBase,                              // lds_addr
                       (unsigned)g,                          // global_addr[31:0]
                       (unsigned)((g >> 32) & 0x01FFFFFFu) | 0x80000000u }; // +type=2
      __builtin_amdgcn_tensor_load_to_lds(g0, g1, gz4, gz4, gz8, 0);
    }
    // ---- fill A slab: gather concatenated [nf|tf|ef|pad] segment, fp32->bf16
    {
      const int kbase = ch * 32 + half;
      __builtin_prefetch(nfR + ((kbase + 32) & (D_ - 1)), 0, 1);  // global_prefetch_b8
      #pragma unroll
      for (int j = 0; j < 16; ++j) {
        int k = kbase + j;
        float v;
        if (k < D_)                v = nfR[k];
        else if (k < D_ + T_)      v = tfR[k - D_];
        else if (k < D_ + T_ + E_) v = efR[k - (D_ + T_)];
        else                       v = 0.f;
        Asl[r * 32 + half + j] = f2bf(v);
      }
    }
    if (t < 32) __builtin_amdgcn_s_wait_tensorcnt(0);
    __syncthreads();

    // ---- A fragment for this wave's 16-row tile
    // (ISA 7.12.2, 16-bit A 16x32): lane<16 -> row=lane, halves
    // {k0..k0+7, k0+16..k0+23} with k0=0; lane>=16 -> same row, k0=8.
    FragU fa;
    {
      const int arow = wave * 16 + (lane & 15);
      const int hb   = (lane >= 16) ? 8 : 0;
      const uint4* p = (const uint4*)&Asl[arow * 32 + hb];
      fa.q[0] = p[0];      // halves k0..k0+7
      fa.q[1] = p[2];      // halves k0+16..k0+23
    }
    // ---- 16 column tiles, software-pipelined B-fragment loads
    const int bcol0 = lane & 15;
    const int hb    = (lane >= 16) ? 8 : 0;
    FragU fb_cur, fb_nxt;
    {
      const uint4* p = (const uint4*)&Bsl[bcol0 * 32 + hb];
      fb_cur.q[0] = p[0];
      fb_cur.q[1] = p[2];
    }
    #pragma unroll
    for (int ct = 0; ct < 16; ++ct) {
      if (ct < 15) {
        const uint4* p = (const uint4*)&Bsl[((ct + 1) * 16 + bcol0) * 32 + hb];
        fb_nxt.q[0] = p[0];
        fb_nxt.q[1] = p[2];
      }
      acc[ct] = __builtin_amdgcn_wmma_f32_16x16x32_bf16(
          false, fa.v, false, fb_cur.v, (short)0, acc[ct], false, false);
      fb_cur = fb_nxt;
    }
  }

  // ---- epilogue: bias + fp32 residual + padding mask, then sum groups of 20.
  // C/D layout: VGPR i, lane<16 -> (M=i, N=lane); lane>=16 -> (M=i+8, N=lane-16).
  const int c   = lane & 15;
  const int rlo = wave * 16 + ((lane >> 4) << 3);
  bool m8[8];
  #pragma unroll
  for (int i = 0; i < 8; ++i) m8[i] = (mask[rowBlk + rlo + i] != 0);

  for (int ct = 0; ct < 16; ++ct) {
    __syncthreads();
    const int col = ct * 16 + c;
    const float bb = bnbr[col];
    #pragma unroll
    for (int i = 0; i < 8; ++i) {
      const int rl = rlo + i;
      const long Rr = rowBlk + rl;
      float v = acc[ct][i] + bb + nf[Rr * D_ + col];
      if (m8[i]) v = 0.f;
      Esl[rl * 16 + c] = v;
    }
    __syncthreads();
    if (t < 128) {                       // 8 groups x 16 cols
      const int g = t >> 4, cc = t & 15;
      float s = 0.f;
      #pragma unroll
      for (int j = 0; j < NNB; ++j) s += Esl[(g * NNB + j) * 16 + cc];
      outSum[((long)blockIdx.x * 8 + g) * D_ + ct * 16 + cc] = s;
    }
  }
}

// ---------------------------------------------------------------------------
// Kernel 2: source projection GEMM  [16384,356]@[356,256] + bias.
// Block = 256 threads = 8 waves, 128 rows.  B slab filled with
// GLOBAL_LOAD_ASYNC_TO_LDS_B128 (ASYNCcnt path).
// ---------------------------------------------------------------------------
__global__ __launch_bounds__(256)
void src_gemm(const float* __restrict__ sf, const float* __restrict__ stf,
              const unsigned short* __restrict__ WsT,
              const float* __restrict__ bsrc,
              float* __restrict__ outSrc /* [B,256] */) {
  __shared__ __align__(16) unsigned short Asl[128 * 32];
  __shared__ __align__(16) unsigned short Bsl[256 * 32];

  const int t    = threadIdx.x;
  const int lane = t & 31;
  const int wave = t >> 5;
  const long rowBlk = (long)blockIdx.x * 128;

  v8f acc[16] = {};

  const int r    = t >> 1;          // 0..127
  const int half = (t & 1) * 16;
  const long R   = rowBlk + r;
  const float* sfR = sf  + R * D_;
  const float* stR = stf + R * T_;

  const unsigned bslBase = lds_addr32(&Bsl[0]);
  const unsigned long long wsBase = (unsigned long long)WsT;

  for (int ch = 0; ch < NCH_SRC; ++ch) {
    __syncthreads();
    // ---- async B-slab fill: 1024 x 16B global->LDS, no VGPR round-trip
    for (int i = t; i < 1024; i += 256) {
      const int c  = i >> 2;
      const int pt = i & 3;
      const unsigned lds = bslBase + (unsigned)(i * 16);
      const unsigned long long g =
          wsBase + (unsigned long long)(c * (KC_SRC * 2) + ch * 64 + pt * 16);
      asm volatile("global_load_async_to_lds_b128 %0, %1, off"
                   :: "v"(lds), "v"(g) : "memory");
    }
    {
      const int kbase = ch * 32 + half;
      #pragma unroll
      for (int j = 0; j < 16; ++j) {
        int k = kbase + j;
        float v;
        if (k < D_)           v = sfR[k];
        else if (k < D_ + T_) v = stR[k - D_];
        else                  v = 0.f;
        Asl[r * 32 + half + j] = f2bf(v);
      }
    }
    asm volatile("s_wait_asynccnt 0x0" ::: "memory");
    __syncthreads();

    FragU fa;
    {
      const int arow = wave * 16 + (lane & 15);
      const int hb   = (lane >= 16) ? 8 : 0;
      const uint4* p = (const uint4*)&Asl[arow * 32 + hb];
      fa.q[0] = p[0];
      fa.q[1] = p[2];
    }
    const int bcol0 = lane & 15;
    const int hb    = (lane >= 16) ? 8 : 0;
    FragU fb_cur, fb_nxt;
    {
      const uint4* p = (const uint4*)&Bsl[bcol0 * 32 + hb];
      fb_cur.q[0] = p[0];
      fb_cur.q[1] = p[2];
    }
    #pragma unroll
    for (int ct = 0; ct < 16; ++ct) {
      if (ct < 15) {
        const uint4* p = (const uint4*)&Bsl[((ct + 1) * 16 + bcol0) * 32 + hb];
        fb_nxt.q[0] = p[0];
        fb_nxt.q[1] = p[2];
      }
      acc[ct] = __builtin_amdgcn_wmma_f32_16x16x32_bf16(
          false, fa.v, false, fb_cur.v, (short)0, acc[ct], false, false);
      fb_cur = fb_nxt;
    }
  }

  const int c   = lane & 15;
  const int rlo = wave * 16 + ((lane >> 4) << 3);
  #pragma unroll
  for (int ct = 0; ct < 16; ++ct) {
    const int col = ct * 16 + c;
    const float bb = bsrc[col];
    #pragma unroll
    for (int i = 0; i < 8; ++i) {
      const long Rr = rowBlk + rlo + i;
      outSrc[Rr * D_ + col] = acc[ct][i] + bb;
    }
  }
}

// ---------------------------------------------------------------------------
// Kernel 3: 2-hop attention + final add.  One wave32 per batch row.
// ---------------------------------------------------------------------------
__global__ __launch_bounds__(256)
void att_kernel(const float* __restrict__ ns, const float* __restrict__ nb,
                const float* __restrict__ Wa, const float* __restrict__ ba,
                float* __restrict__ out) {
  const int wave = threadIdx.x >> 5, lane = threadIdx.x & 31;
  const long b = (long)blockIdx.x * 8 + wave;
  const float* s  = ns + b * D_;
  const float* n0 = nb + (b * 2 + 0) * D_;
  const float* n1 = nb + (b * 2 + 1) * D_;

  float sv[8], v0[8], v1[8];
  float es = 0.f, e0 = 0.f, e1 = 0.f;
  #pragma unroll
  for (int j = 0; j < 8; ++j) {
    const int c = j * 32 + lane;
    sv[j] = s[c]; v0[j] = n0[c]; v1[j] = n1[c];
    es += sv[j] * Wa[c];
    e0 += v0[j] * Wa[D_ + c];
    e1 += v1[j] * Wa[D_ + c];
  }
  #pragma unroll
  for (int off = 16; off > 0; off >>= 1) {
    es += __shfl_xor(es, off, 32);
    e0 += __shfl_xor(e0, off, 32);
    e1 += __shfl_xor(e1, off, 32);
  }
  const float bb = ba[0];
  const float a0 = es + e0 + bb, a1 = es + e1 + bb;
  const float m  = fmaxf(a0, a1);
  float w0 = __expf(a0 - m), w1 = __expf(a1 - m);
  const float inv = 1.f / (w0 + w1);
  w0 *= inv; w1 *= inv;
  #pragma unroll
  for (int j = 0; j < 8; ++j) {
    const int c = j * 32 + lane;
    out[b * D_ + c] = sv[j] + w0 * v0[j] + w1 * v1[j];
  }
}

// ---------------------------------------------------------------------------
// Host launcher.  Workspace layout (bytes):
//   [0, 278528)                       WnT  bf16 [256][544]
//   [278528, 475136)                  WsT  bf16 [256][384]
//   [475136, 17252352)                new_src  f32 [16384][256]
//   [17252352, 50806784)              nbr_sum  f32 [16384*2][256]
// ---------------------------------------------------------------------------
extern "C" void kernel_launch(void* const* d_in, const int* in_sizes, int n_in,
                              void* d_out, int out_size, void* d_ws, size_t ws_size,
                              hipStream_t stream) {
  (void)in_sizes; (void)n_in; (void)out_size; (void)ws_size;
  const float* src_nf = (const float*)d_in[0];
  const float* src_tf = (const float*)d_in[1];
  const float* nbr_f  = (const float*)d_in[2];
  const float* nbr_tf = (const float*)d_in[3];
  const float* edge_f = (const float*)d_in[4];
  const unsigned char* mask = (const unsigned char*)d_in[5];
  const float* Wn = (const float*)d_in[6];
  const float* bn = (const float*)d_in[7];
  const float* Wsp = (const float*)d_in[8];
  const float* bs = (const float*)d_in[9];
  const float* Wa = (const float*)d_in[10];
  const float* ba = (const float*)d_in[11];

  char* ws = (char*)d_ws;
  unsigned short* WnT = (unsigned short*)(ws);
  unsigned short* WsT = (unsigned short*)(ws + 278528);
  float* newSrc = (float*)(ws + 475136);
  float* nbrSum = (float*)(ws + 17252352);

  const int nPrep = D_ * KC_NBR + D_ * KC_SRC;   // 237568
  prep_weights<<<(nPrep + 255) / 256, 256, 0, stream>>>(Wn, Wsp, WnT, WsT);
  nbr_gemm<<<ROWS_NBR / 160, 320, 0, stream>>>(nbr_f, nbr_tf, edge_f, mask,
                                               WnT, bn, nbrSum);
  src_gemm<<<B_ / 128, 256, 0, stream>>>(src_nf, src_tf, WsT, bs, newSrc);
  att_kernel<<<B_ / 8, 256, 0, stream>>>(newSrc, nbrSum, Wa, ba, (float*)d_out);
}
